// DynSIHAAttention_28312424415699
// MI455X (gfx1250) — matrix operand order repaired and checked by reading, not compiled, not run on hardware
//
#include <hip/hip_runtime.h>
#include <hip/hip_bf16.h>

// Problem constants (match reference)
constexpr int kB = 4;
constexpr int kT = 1024;
constexpr int kC = 1024;
constexpr int kH = 16;
constexpr int kD = 64;
constexpr int kE = 8;
constexpr int kHID = 128;

typedef _Float16 h8   __attribute__((ext_vector_type(8)));
typedef _Float16 v16h __attribute__((ext_vector_type(16)));
typedef float    v8f  __attribute__((ext_vector_type(8)));
typedef int      vs4i __attribute__((vector_size(16)));   // matches async-LDS builtin param

#define AS1 __attribute__((address_space(1)))
#define AS3 __attribute__((address_space(3)))

union FragA {
    v16h v;
    h8   h[2];
    _Float16 e[16];
};

// ---------------------------------------------------------------------------
// DPP16 butterfly reductions (VALU-only, no LDS pipe).
// Masks {1,2,7,15} span GF(2)^4 -> full 16-lane-group reduction.
// ---------------------------------------------------------------------------
template <int CTRL>
__device__ __forceinline__ float dpp_xf(float x) {
    return __int_as_float(__builtin_amdgcn_update_dpp(
        0, __float_as_int(x), CTRL, 0xF, 0xF, true));
}
__device__ __forceinline__ float red16_max(float x) {
    x = fmaxf(x, dpp_xf<0xB1>(x));   // xor1  (quad_perm 1,0,3,2)
    x = fmaxf(x, dpp_xf<0x4E>(x));   // xor2  (quad_perm 2,3,0,1)
    x = fmaxf(x, dpp_xf<0x141>(x));  // xor7  (row_half_mirror)
    x = fmaxf(x, dpp_xf<0x140>(x));  // xor15 (row_mirror)
    return x;
}
__device__ __forceinline__ float red16_sum(float x) {
    x += dpp_xf<0xB1>(x);
    x += dpp_xf<0x4E>(x);
    x += dpp_xf<0x141>(x);
    x += dpp_xf<0x140>(x);
    return x;
}
__device__ __forceinline__ float wave_sum32(float x) {
    x = red16_sum(x);
    x += __shfl_xor(x, 16, 32);   // single cross-half exchange
    return x;
}
__device__ __forceinline__ float gelu_tanh(float x) {
    float x3 = x * x * x;
    return 0.5f * x * (1.0f + tanhf(0.7978845608028654f * (x + 0.044715f * x3)));
}

// CDNA5 async global->LDS staging path (codegen-confirmed in round 3)
#if defined(__AMDGCN__) && defined(__gfx1250__) && \
    __has_builtin(__builtin_amdgcn_global_load_async_to_lds_b128) && \
    __has_builtin(__builtin_amdgcn_s_wait_asynccnt)
#define USE_ASYNC_LDS 1
#else
#define USE_ASYNC_LDS 0
#endif

// ---------------------------------------------------------------------------
// Kernel 0: convert o_w (fp32 [C,C]) -> f16
// ---------------------------------------------------------------------------
__global__ __launch_bounds__(256) void convert_ow(const float* __restrict__ w,
                                                  _Float16* __restrict__ w16) {
    int i = blockIdx.x * 256 + threadIdx.x;
    w16[i] = (_Float16)w[i];
}

// ---------------------------------------------------------------------------
// Kernel 1: per-head top-2 MoE projection for q/k/v + RoPE.
// One wave (32 lanes) per (b,t,h). q,k -> [B,H,T,D] f16; v -> [B,H,D,T] f16.
// ---------------------------------------------------------------------------
__global__ __launch_bounds__(256) void moe_qkv_rope(
    const float* __restrict__ x, const int* __restrict__ pos_ids,
    const float* __restrict__ cosT, const float* __restrict__ sinT,
    const float* __restrict__ rq, const float* __restrict__ rk, const float* __restrict__ rv,
    const float* __restrict__ qw1, const float* __restrict__ qw2,
    const float* __restrict__ kw1, const float* __restrict__ kw2,
    const float* __restrict__ vw1, const float* __restrict__ vw2,
    _Float16* __restrict__ q16, _Float16* __restrict__ k16, _Float16* __restrict__ v16)
{
    __shared__ float sx[8 * kD];     // staged input per wave
    __shared__ float sh[8 * kHID];   // staged hidden per wave

    const int lane  = threadIdx.x & 31;
    const int wslot = threadIdx.x >> 5;
    const int wid   = blockIdx.x * 8 + wslot;   // token-head id
    const int h = wid % kH;
    const int t = (wid / kH) % kT;
    const int b = wid / (kH * kT);

    const float* xrow = x + ((size_t)(b * kT + t)) * kC + h * kD;
    const float x0 = xrow[lane];
    const float x1 = xrow[lane + 32];
    sx[wslot * kD + lane]      = x0;
    sx[wslot * kD + lane + 32] = x1;
    __syncthreads();

    const int pos = pos_ids[b * kT + t];
    const float c0 = cosT[pos * kD + lane], c1 = cosT[pos * kD + lane + 32];
    const float s0 = sinT[pos * kD + lane], s1 = sinT[pos * kD + lane + 32];

    const float* routers[3] = {rq, rk, rv};
    const float* W1[3] = {qw1, kw1, vw1};
    const float* W2[3] = {qw2, kw2, vw2};

    for (int p = 0; p < 3; ++p) {
        // ---- router logits (wave-wide dot products, DPP reduction) ----
        float logits[kE];
        const float* r = routers[p];
        #pragma unroll
        for (int e = 0; e < kE; ++e) {
            float part = x0 * r[lane * kE + e] + x1 * r[(lane + 32) * kE + e];
            logits[e] = wave_sum32(part);
        }
        // ---- top-2 + softmax weights ----
        float v1m = logits[0]; int i1 = 0;
        #pragma unroll
        for (int e = 1; e < kE; ++e) if (logits[e] > v1m) { v1m = logits[e]; i1 = e; }
        float v2m = -3.0e38f; int i2 = 0;
        #pragma unroll
        for (int e = 0; e < kE; ++e) if (e != i1 && logits[e] > v2m) { v2m = logits[e]; i2 = e; }
        const float wa = 1.0f / (1.0f + __expf(v2m - v1m));
        const float wb = 1.0f - wa;

        const int   experts[2] = {i1, i2};
        const float wgt[2]     = {wa, wb};
        float acc0 = 0.f, acc1 = 0.f;

        for (int kk = 0; kk < 2; ++kk) {
            const int ee = experts[kk];
            const float* w1 = W1[p] + (size_t)ee * kD * kHID;   // [D,HID]
            const float* w2 = W2[p] + (size_t)ee * kHID * kD;   // [HID,D]

            // hidden = gelu(x @ w1): lane covers f = lane + 32*j
            float hv0 = 0.f, hv1 = 0.f, hv2 = 0.f, hv3 = 0.f;
            for (int d = 0; d < kD; ++d) {
                const float xv = sx[wslot * kD + d];
                const float* wr = w1 + d * kHID;
                hv0 += xv * wr[lane];
                hv1 += xv * wr[lane + 32];
                hv2 += xv * wr[lane + 64];
                hv3 += xv * wr[lane + 96];
            }
            sh[wslot * kHID + lane]      = gelu_tanh(hv0);
            sh[wslot * kHID + lane + 32] = gelu_tanh(hv1);
            sh[wslot * kHID + lane + 64] = gelu_tanh(hv2);
            sh[wslot * kHID + lane + 96] = gelu_tanh(hv3);
            __syncthreads();

            // out += wgt * (hidden @ w2): lane covers g = lane, lane+32
            float o0 = 0.f, o1 = 0.f;
            for (int f = 0; f < kHID; ++f) {
                const float hval = sh[wslot * kHID + f];
                const float* wr = w2 + f * kD;
                o0 += hval * wr[lane];
                o1 += hval * wr[lane + 32];
            }
            acc0 += wgt[kk] * o0;
            acc1 += wgt[kk] * o1;
            __syncthreads();
        }

        if (p == 2) {
            // V stored transposed: [B,H,D,T]
            v16[(((size_t)(b * kH + h)) * kD + lane) * kT + t]      = (_Float16)acc0;
            v16[(((size_t)(b * kH + h)) * kD + lane + 32) * kT + t] = (_Float16)acc1;
        } else {
            // RoPE
            const float r0 = acc0 * c0 - acc1 * s0;
            const float r1 = acc1 * c1 + acc0 * s1;
            _Float16* dst = (p == 0) ? q16 : k16;
            const size_t base = (((size_t)(b * kH + h)) * kT + t) * kD;
            dst[base + lane]      = (_Float16)r0;
            dst[base + lane + 32] = (_Float16)r1;
        }
    }
}

// ---------------------------------------------------------------------------
// Kernel 2: causal flash attention with WMMA f16.
// grid = (T/128, B*H). 8 waves/block; wave owns a 16-row Q tile.
// Double-buffered K/V tiles in LDS; next tile's async-to-LDS DMA is issued
// before computing the current tile so the DMA overlaps the WMMAs, with the
// ASYNCcnt wait deferred to the next iteration top.
// ---------------------------------------------------------------------------
__global__ __launch_bounds__(256) void flash_attn(
    const _Float16* __restrict__ q16, const _Float16* __restrict__ k16,
    const _Float16* __restrict__ v16, _Float16* __restrict__ o16)
{
    __shared__ __align__(16) _Float16 skbuf[2][32 * 64];  // K tiles: 32 keys x 64 d
    __shared__ __align__(16) _Float16 svbuf[2][64 * 32];  // V tiles: 64 d x 32 keys
    __shared__ __align__(16) _Float16 sp[8 * 512];        // per-wave P tile (16x32)

    const int tid   = threadIdx.x;
    const int lane  = tid & 31;
    const int wslot = tid >> 5;
    const int bh = blockIdx.y;                 // b*H + h
    const int b = bh / kH, h = bh % kH;
    const int qbase = blockIdx.x * 128;
    const int q0 = qbase + wslot * 16;
    const int lm = lane & 15;
    const int hi = (lane < 16) ? 0 : 1;

    const _Float16* kgbase = k16 + ((size_t)bh) * kT * kD;   // [T,D]
    const _Float16* vgbase = v16 + ((size_t)bh) * kD * kT;   // [D,T]

    // cooperative staging indices (one 16B chunk per thread per tile)
    const int skey = tid >> 3, skp = tid & 7;   // K: key row, 8-half part
    const int svd  = tid >> 2, svp = tid & 3;   // V: d row, 8-half part

    auto stage = [&](int kv0, int buf) {
        const _Float16* kg = kgbase + ((size_t)(kv0 + skey)) * kD + skp * 8;
        const _Float16* vg = vgbase + (size_t)svd * kT + kv0 + svp * 8;
        _Float16* kl = &skbuf[buf][skey * 64 + skp * 8];
        _Float16* vl = &svbuf[buf][svd * 32 + svp * 8];
#if USE_ASYNC_LDS
        __builtin_amdgcn_global_load_async_to_lds_b128(
            (AS1 vs4i*)kg, (AS3 vs4i*)kl, 0, 0);
        __builtin_amdgcn_global_load_async_to_lds_b128(
            (AS1 vs4i*)vg, (AS3 vs4i*)vl, 0, 0);
#else
        *(h8*)kl = *(const h8*)kg;
        *(h8*)vl = *(const h8*)vg;
#endif
    };

    // ---- Q A-fragments for the two 32-wide D chunks (held in registers) ----
    FragA qa[2];
    {
        const _Float16* qrow = q16 + (((size_t)bh) * kT + (q0 + lm)) * kD;
        const int kb = hi ? 16 : 0;
        #pragma unroll
        for (int c = 0; c < 2; ++c) {
            qa[c].h[0] = *(const h8*)(qrow + c * 32 + kb);
            qa[c].h[1] = *(const h8*)(qrow + c * 32 + kb + 8);
        }
    }

    v8f Oacc[4] = {};
    float mrow[8], lrow[8];
    #pragma unroll
    for (int r = 0; r < 8; ++r) { mrow[r] = -3.0e38f; lrow[r] = 0.f; }

    const int kend = qbase + 128;      // block-uniform causal bound
    const int ntiles = kend / 32;      // >= 4
    const float scale = 0.125f;        // 1/sqrt(64)

    stage(0, 0);                       // prologue: tile 0 -> buffer 0

    #pragma unroll 2
    for (int it = 0; it < ntiles; ++it) {
        const int kv0 = it * 32;
        const int cur = it & 1;
#if USE_ASYNC_LDS
        __builtin_amdgcn_s_wait_asynccnt(0);   // tile `it` landed (this wave's part)
#endif
        __syncthreads();                        // all waves' parts visible

        // issue next tile's DMA into the other buffer; overlaps with compute
        if (it + 1 < ntiles) stage(kv0 + 32, cur ^ 1);

        const _Float16* sk = skbuf[cur];
        const _Float16* sv = svbuf[cur];

        // ---- S = Q K^T for two 16-key subtiles (B-frags from LDS) ----
        v8f s[2];
        #pragma unroll
        for (int ss = 0; ss < 2; ++ss) {
            const _Float16* krow = sk + (ss * 16 + lm) * 64;
            v8f acc = {};
            #pragma unroll
            for (int c = 0; c < 2; ++c) {
                const int off = c * 32 + (hi ? 16 : 0);
                FragA kbf;
                kbf.h[0] = *(const h8*)(krow + off);
                kbf.h[1] = *(const h8*)(krow + off + 8);
                acc = __builtin_amdgcn_wmma_f32_16x16x32_f16(
                    false, qa[c].v, false, kbf.v, (short)0, acc, false, false);
            }
            s[ss] = acc;
        }

        // ---- causal mask + online softmax (DPP16 reductions, VALU only) ----
        float p0[8], p1[8], alpha[8];
        #pragma unroll
        for (int r = 0; r < 8; ++r) {
            const int qg = q0 + r + hi * 8;
            float v0 = s[0][r] * scale; if (kv0 + lm > qg)      v0 = -1.0e30f;
            float v1 = s[1][r] * scale; if (kv0 + 16 + lm > qg) v1 = -1.0e30f;
            const float tmax = red16_max(fmaxf(v0, v1));
            const float mnew = fmaxf(mrow[r], tmax);
            const float a  = __expf(mrow[r] - mnew);
            const float e0 = __expf(v0 - mnew);
            const float e1 = __expf(v1 - mnew);
            lrow[r] = lrow[r] * a + red16_sum(e0 + e1);
            mrow[r] = mnew;
            alpha[r] = a;
            p0[r] = e0; p1[r] = e1;
        }
        #pragma unroll
        for (int c = 0; c < 4; ++c) {
            #pragma unroll
            for (int r = 0; r < 8; ++r) Oacc[c][r] *= alpha[r];
        }

        // ---- spill P (C-layout) to LDS, reload as A-fragment ----
        _Float16* pw = sp + wslot * 512;
        #pragma unroll
        for (int r = 0; r < 8; ++r) {
            const int m = r + hi * 8;
            pw[m * 32 + lm]      = (_Float16)p0[r];
            pw[m * 32 + 16 + lm] = (_Float16)p1[r];
        }
        __syncthreads();
        FragA pa;
        {
            const int kb = hi ? 16 : 0;
            pa.h[0] = *(const h8*)(pw + lm * 32 + kb);
            pa.h[1] = *(const h8*)(pw + lm * 32 + kb + 8);
        }

        // ---- O += P @ V (V B-frags from LDS, contiguous along keys) ----
        const int vkb = hi ? 16 : 0;
        #pragma unroll
        for (int c = 0; c < 4; ++c) {
            const _Float16* vrow = sv + (c * 16 + lm) * 32 + vkb;
            FragA vb;
            vb.h[0] = *(const h8*)(vrow);
            vb.h[1] = *(const h8*)(vrow + 8);
            Oacc[c] = __builtin_amdgcn_wmma_f32_16x16x32_f16(
                false, pa.v, false, vb.v, (short)0, Oacc[c], false, false);
        }
        // no trailing barrier needed: buffer cur is only overwritten by the
        // stage() issued after the *next* iteration's top barrier.
    }

    // ---- normalize and store to [B,T, H*D] f16 ----
    #pragma unroll
    for (int r = 0; r < 8; ++r) {
        const float inv = 1.0f / lrow[r];
        const int qg = q0 + r + hi * 8;
        _Float16* orow = o16 + ((size_t)(b * kT + qg)) * kC + h * kD;
        #pragma unroll
        for (int c = 0; c < 4; ++c) {
            orow[c * 16 + lm] = (_Float16)(Oacc[c][r] * inv);
        }
    }
}

// ---------------------------------------------------------------------------
// Kernel 3: out = X_f16 [B*T, C] @ o_w^T, fp32 result.
// grid = ((B*T)/128, C/32); 8 waves/block; each wave computes a 16x32 tile
// (one A-frag feeds two WMMAs -> halved A traffic).
// ---------------------------------------------------------------------------
__global__ __launch_bounds__(256) void out_proj(
    const _Float16* __restrict__ x16, const _Float16* __restrict__ w16,
    float* __restrict__ out)
{
    const int lane  = threadIdx.x & 31;
    const int wslot = threadIdx.x >> 5;
    const int lm = lane & 15;
    const int m0 = blockIdx.x * 128 + wslot * 16;
    const int g0 = blockIdx.y * 32;

    const _Float16* arow  = x16 + ((size_t)(m0 + lm)) * kC;
    const _Float16* brow0 = w16 + ((size_t)(g0 + lm)) * kC;
    const _Float16* brow1 = w16 + ((size_t)(g0 + 16 + lm)) * kC;
    const int akb = (lane < 16) ? 0 : 16;
    const int bkb = (lane < 16) ? 0 : 16;

    v8f acc0 = {}, acc1 = {};
    #pragma unroll 4
    for (int ct = 0; ct < kC; ct += 32) {
        __builtin_prefetch(arow + ct + 512, 0, 0);
        __builtin_prefetch(brow0 + ct + 512, 0, 0);
        __builtin_prefetch(brow1 + ct + 512, 0, 0);
        FragA a, bf0, bf1;
        a.h[0]   = *(const h8*)(arow + ct + akb);
        a.h[1]   = *(const h8*)(arow + ct + akb + 8);
        bf0.h[0] = *(const h8*)(brow0 + ct + bkb);
        bf0.h[1] = *(const h8*)(brow0 + ct + bkb + 8);
        bf1.h[0] = *(const h8*)(brow1 + ct + bkb);
        bf1.h[1] = *(const h8*)(brow1 + ct + bkb + 8);
        acc0 = __builtin_amdgcn_wmma_f32_16x16x32_f16(
            false, a.v, false, bf0.v, (short)0, acc0, false, false);
        acc1 = __builtin_amdgcn_wmma_f32_16x16x32_f16(
            false, a.v, false, bf1.v, (short)0, acc1, false, false);
    }

    const int hig = (lane < 16) ? 0 : 8;
    #pragma unroll
    for (int r = 0; r < 8; ++r) {
        out[((size_t)(m0 + r + hig)) * kC + g0 + lm]      = acc0[r];
        out[((size_t)(m0 + r + hig)) * kC + g0 + 16 + lm] = acc1[r];
    }
}

// ---------------------------------------------------------------------------
extern "C" void kernel_launch(void* const* d_in, const int* in_sizes, int n_in,
                              void* d_out, int out_size, void* d_ws, size_t ws_size,
                              hipStream_t stream) {
    (void)in_sizes; (void)n_in; (void)out_size; (void)ws_size;

    const float* x    = (const float*)d_in[0];
    const int*   pos  = (const int*)d_in[1];
    const float* cosT = (const float*)d_in[2];
    const float* sinT = (const float*)d_in[3];
    const float* rq   = (const float*)d_in[4];
    const float* rk   = (const float*)d_in[5];
    const float* rv   = (const float*)d_in[6];
    const float* qw1  = (const float*)d_in[7];
    const float* qw2  = (const float*)d_in[8];
    const float* kw1  = (const float*)d_in[9];
    const float* kw2  = (const float*)d_in[10];
    const float* vw1  = (const float*)d_in[11];
    const float* vw2  = (const float*)d_in[12];
    const float* ow   = (const float*)d_in[13];

    char* ws = (char*)d_ws;
    const size_t MB = 1024 * 1024;
    _Float16* q16 = (_Float16*)(ws);                 //  8 MB [B,H,T,D]
    _Float16* k16 = (_Float16*)(ws + 8  * MB);       //  8 MB [B,H,T,D]
    _Float16* v16 = (_Float16*)(ws + 16 * MB);       //  8 MB [B,H,D,T]
    _Float16* o16 = (_Float16*)(ws + 24 * MB);       //  8 MB [B,T,C]
    _Float16* w16 = (_Float16*)(ws + 32 * MB);       //  2 MB [C,C]

    convert_ow<<<(kC * kC) / 256, 256, 0, stream>>>(ow, w16);
    moe_qkv_rope<<<(kB * kT * kH) / 8, 256, 0, stream>>>(
        x, pos, cosT, sinT, rq, rk, rv, qw1, qw2, kw1, kw2, vw1, vw2, q16, k16, v16);
    dim3 g2(kT / 128, kB * kH);
    flash_attn<<<g2, 256, 0, stream>>>(q16, k16, v16, o16);
    dim3 g3((kB * kT) / 128, kC / 32);
    out_proj<<<g3, 256, 0, stream>>>(o16, w16, (float*)d_out);
}
